// GNN_44513041056055
// MI455X (gfx1250) — matrix operand order repaired and checked by reading, not compiled, not run on hardware
//
#include <hip/hip_runtime.h>
#include <cstdint>
#include <cstddef>

typedef _Float16 f16_t;
typedef __attribute__((ext_vector_type(16))) _Float16 v16h;
typedef __attribute__((ext_vector_type(8)))  float    v8f;

#define DEV __device__ __forceinline__

static constexpr int NN   = 2000;   // nodes
static constexpr int EE   = 16000;  // edges
static constexpr int H    = 8;      // heads
static constexpr int HD   = 32;     // head dim
static constexpr int W    = 6;      // time steps
static constexpr int LAM  = 31;
static constexpr int NB   = 8;
static constexpr int PP   = 20;     // particles
static constexpr int HH   = H * HD;   // 256
static constexpr int G4   = 4 * HH;   // 1024
static constexpr float PI2 = 6.28318530717958647692f;

DEV float sigm(float x)  { return 1.f / (1.f + __expf(-x)); }
DEV float lrelu(float x) { return x > 0.f ? x : 0.01f * x; }
DEV float wsum(float v) {
#pragma unroll
  for (int m = 16; m; m >>= 1) v += __shfl_xor(v, m, 32);
  return v;
}

// ---------------------------------------------------------------------------
// WMMA GEMM:  C[M,N] = A[M,K](f16,row) x BT[N,K](f16,row)^T (+bias[N]) (+add)
//
// Block = 8 waves, tile 128(M) x 64(N).  Each wave owns one 16-row strip and
// 4 N-tiles (16x64) -> A fragment loaded once per k-step, reused by 4 WMMAs.
// B slab (64 cols x 32 K = 4KB) is cooperatively staged to LDS (one b128 per
// thread -> ds_store_b128), double-buffered, shared by all 8 waves; fragments
// come from ds_load_b128 pairs.  One s_barrier per k-step.  No wave exits
// early (M-tiles clamped, stores guarded) so barriers are safe.
//
// Fragment layouts per CDNA5 ISA 7.12.2 (V_WMMA_F32_16X16X32_F16):
//   A 16x32 f16: lane = M%16 + 16*kh; VGPR0-3 K=kh*8+0..7, VGPR4-7 K=+16
//   B 32x16 f16: lane = N%16 + 16*kh; VGPR0-3 K=kh*8+0..7, VGPR4-7 K=+16 (same as A,
//   two ds_load_b128 per fragment)
// ---------------------------------------------------------------------------
struct __attribute__((aligned(16))) U4 { unsigned x, y, z, w; };
union AFrag { v16h v; U4 q[2]; };

static constexpr int BN = 64;    // block N tile
static constexpr int BM = 128;   // block M tile (8 waves x 16)

__global__ void k_wmma_gemm(const f16_t* __restrict__ A, const f16_t* __restrict__ BT,
                            const float* __restrict__ bias, const float* __restrict__ add,
                            float* __restrict__ C, int M, int N, int K,
                            long long sA, long long sB, long long sC, int flags) {
  __shared__ __align__(32) f16_t lb[2][BN * 32];   // [buf][col*32 + k']
  const int z = blockIdx.z;
  A  += (long long)z * sA;
  BT += (long long)z * sB;
  C  += (long long)z * sC;
  const float* addp = (flags & 2) ? (add + (long long)z * sC) : nullptr;

  const int tid  = threadIdx.x;
  const int wid  = tid >> 5;
  const int lane = tid & 31;
  const int tn0  = blockIdx.x * BN;
  int tm = blockIdx.y * BM + wid * 16;
  const bool mvalid = (tm < M);
  if (tm > M - 16) tm = M - 16;          // clamp (M >= 16, multiple of 16)

  const int mr = lane & 15;
  const int kh = lane >> 4;
  const f16_t* aB = A + (size_t)(tm + mr) * K + kh * 8;

  // B staging: thread copies 8 halves (16B) of one BT row into LDS
  const int scol  = tid >> 2;            // 0..63  (column within slab)
  const int squad = (tid & 3) * 8;       // 0,8,16,24 (K offset)
  const int sOff  = scol * 32 + squad;   // element offset within one buffer
  int gcol = tn0 + scol;
  if (gcol > N - 1) gcol = N - 1;        // clamp OOB columns (stores guarded)
  const f16_t* bSrc = BT + (size_t)gcol * K + squad;

  v8f acc[4] = { {}, {}, {}, {} };

  *(U4*)(&lb[0][sOff]) = *(const U4*)(bSrc);   // stage k-step 0 (ds_store_b128)
  __syncthreads();

  int buf = 0;
  for (int kb = 0; kb < K; kb += 32) {
    if (kb + 32 < K) {                   // prefetch next slab into other buffer
      U4 t = *(const U4*)(bSrc + kb + 32);
      *(U4*)(&lb[buf ^ 1][sOff]) = t;    // direct LDS index -> ds_store_b128
    }

    AFrag af;
    af.q[0] = *(const U4*)(aB + kb);
    af.q[1] = *(const U4*)(aB + kb + 16);

#pragma unroll
    for (int nt = 0; nt < 4; ++nt) {
      AFrag bf;
      bf.q[0] = *(const U4*)(&lb[buf][(nt * 16 + mr) * 32 + kh * 8]);
      bf.q[1] = *(const U4*)(&lb[buf][(nt * 16 + mr) * 32 + 16 + kh * 8]);
      acc[nt] = __builtin_amdgcn_wmma_f32_16x16x32_f16(false, af.v, false, bf.v,
                                                       (short)0, acc[nt], false, false);
    }
    __syncthreads();                     // all reads of buf & writes of buf^1 done
    buf ^= 1;
  }

  const int r0 = tm + kh * 8;
#pragma unroll
  for (int nt = 0; nt < 4; ++nt) {
    const int cb = tn0 + nt * 16;
    if (!mvalid || cb >= N) continue;    // wave-uniform guards
    const int col = cb + mr;
    const float bb = (flags & 1) ? bias[col] : 0.f;
#pragma unroll
    for (int r = 0; r < 8; ++r) {
      size_t off = (size_t)(r0 + r) * N + col;
      float v = acc[nt][r] + bb;
      if (flags & 2) v += addp[off];
      C[off] = v;
    }
  }
}

// ------------------------------- utilities ---------------------------------
__global__ void k_zero(float* p, long long n) {
  long long i = (long long)blockIdx.x * blockDim.x + threadIdx.x;
  if (i < n) p[i] = 0.f;
}
__global__ void k_copy(const float* __restrict__ s, float* __restrict__ d, long long n) {
  long long i = (long long)blockIdx.x * blockDim.x + threadIdx.x;
  if (i < n) d[i] = s[i];
}
__global__ void k_f32h(const float* __restrict__ s, f16_t* __restrict__ d, long long n) {
  long long i = (long long)blockIdx.x * blockDim.x + threadIdx.x;
  if (i < n) d[i] = (f16_t)s[i];
}
// dst[b][n][k] = src[b][k][n]  (build BT = B^T in f16)
__global__ void k_f32hT(const float* __restrict__ s, f16_t* __restrict__ d,
                        int K, int N, long long total) {
  long long i = (long long)blockIdx.x * blockDim.x + threadIdx.x;
  if (i >= total) return;
  long long kn = (long long)K * N;
  int b  = (int)(i / kn);
  int r  = (int)(i % kn);
  int nn = r / K;
  int kk = r % K;
  d[i] = (f16_t)s[(long long)b * kn + (long long)kk * N + nn];
}
__global__ void k_add2(const float* a, const float* b, float* d, int n) {
  int i = blockIdx.x * blockDim.x + threadIdx.x;
  if (i < n) d[i] = a[i] + b[i];
}

// ------------------------ positional encoding MLP --------------------------
__global__ void k_pe(const float* __restrict__ pos, const float* __restrict__ W1,
                     const float* __restrict__ b1, const float* __restrict__ g1,
                     const float* __restrict__ be1, const float* __restrict__ W2,
                     const float* __restrict__ b2, float* __restrict__ pe) {
  int wid = threadIdx.x >> 5, lane = threadIdx.x & 31;
  int node = blockIdx.x * 8 + wid;
  if (node >= NN) return;
  float h1 = b1[lane];
#pragma unroll
  for (int i = 0; i < 3; ++i) h1 += pos[node * 3 + i] * W1[i * 32 + lane];
  h1 = lrelu(h1);
  float m  = wsum(h1) * (1.f / 32.f);
  float df = h1 - m;
  float var = wsum(df * df) * (1.f / 32.f);
  float hn = df * rsqrtf(var + 1e-5f) * g1[lane] + be1[lane];
  float acc = b2[lane];
  for (int j = 0; j < 32; ++j) acc += __shfl(hn, j, 32) * W2[j * 32 + lane];
  pe[node * 32 + lane] = acc;
}

// ----------------------------- input embedding -----------------------------
__global__ void k_embed(const float* __restrict__ x, const float* __restrict__ tt,
                        const float* __restrict__ Wm, const float* __restrict__ bm,
                        float* __restrict__ xe) {
  int i = blockIdx.x * blockDim.x + threadIdx.x;   // W*NN*32
  if (i >= W * NN * 32) return;
  int d = i & 31;
  int b = (i >> 5) % NN;
  int t = i / (NN * 32);
  float acc = bm[d];
  for (int l = 0; l < LAM; ++l)
    acc += x[b * (W * LAM) + t * LAM + l] * Wm[l * 32 + d];
  for (int l = 0; l < LAM; ++l) {
    float tv = tt[t * LAM + l];
    acc += tv                * Wm[(LAM     + l) * 32 + d];
    acc += __sinf(PI2 * tv)  * Wm[(2 * LAM + l) * 32 + d];
    acc += __cosf(PI2 * tv)  * Wm[(3 * LAM + l) * 32 + d];
  }
  xe[i] = acc;
}

// ------------------------- graph-cell sub-kernels --------------------------
// pack xhc = [xx, h, c] into (H, NN, 96) f16 layout for the per-head GEMM
__global__ void k_pack_xhc(f16_t* __restrict__ xhc, const float* __restrict__ xx,
                           int rs, int hs, const float* __restrict__ hst,
                           const float* __restrict__ cst) {
  int i = blockIdx.x * blockDim.x + threadIdx.x;   // H*NN*96
  if (i >= H * NN * 96) return;
  int j = i % 96;
  int b = (i / 96) % NN;
  int h = i / (NN * 96);
  float v;
  if (j < 32)      v = xx[(size_t)b * rs + h * hs + j];
  else if (j < 64) v = hst[((size_t)b * H + h) * 32 + (j - 32)];
  else             v = cst[((size_t)b * H + h) * 32 + (j - 64)];
  xhc[i] = (f16_t)v;
}

// leaky -> LN over (3,H,HD)=768 per node -> split q,k,v0 (NN,H,HD)
__global__ void k_qkv_norm(const float* __restrict__ traw, const float* __restrict__ gam,
                           const float* __restrict__ bet, float* __restrict__ q,
                           float* __restrict__ k, float* __restrict__ v0) {
  __shared__ float s1[256], s2[256];
  int b = blockIdx.x, tid = threadIdx.x;
  float vals[3]; int gs[3], hs_[3], ds[3];
  float ls = 0.f, lq = 0.f;
#pragma unroll
  for (int r = 0; r < 3; ++r) {
    int j = tid + r * 256;
    int g = j >> 8, rem = j & 255, h = rem >> 5, d = rem & 31;
    float v = lrelu(traw[((size_t)h * NN + b) * 96 + g * 32 + d]);
    vals[r] = v; gs[r] = g; hs_[r] = h; ds[r] = d;
    ls += v; lq += v * v;
  }
  s1[tid] = ls; s2[tid] = lq; __syncthreads();
  for (int st = 128; st; st >>= 1) {
    if (tid < st) { s1[tid] += s1[tid + st]; s2[tid] += s2[tid + st]; }
    __syncthreads();
  }
  float m   = s1[0] * (1.f / 768.f);
  float var = s2[0] * (1.f / 768.f) - m * m;
  float rs  = rsqrtf(var + 1e-5f);
#pragma unroll
  for (int r = 0; r < 3; ++r) {
    int j = tid + r * 256;
    float yn = (vals[r] - m) * rs * gam[j] + bet[j];
    float* dst = gs[r] == 0 ? q : (gs[r] == 1 ? k : v0);
    dst[((size_t)b * H + hs_[r]) * 32 + ds[r]] = yn;
  }
}

// one wave per (edge, head): s = mean_d q[src]*k[dst]; w = exp(s + pe_src - pe_dst)
__global__ void k_attn(const float* __restrict__ q, const float* __restrict__ k,
                       const float* __restrict__ v0, const float* __restrict__ pe,
                       const int* __restrict__ ei, float* __restrict__ num,
                       float* __restrict__ den) {
  int gw   = (blockIdx.x * blockDim.x + threadIdx.x) >> 5;
  int lane = threadIdx.x & 31;
  if (gw >= EE * H) return;
  int e = gw >> 3, h = gw & 7;
  int src = ei[e], dst = ei[EE + e];
  float s = q[((size_t)src * H + h) * 32 + lane] * k[((size_t)dst * H + h) * 32 + lane];
  s = wsum(s) * (1.f / 32.f);
  float w = __expf(s + pe[src * 32 + lane] - pe[dst * 32 + lane]);
  atomicAdd(&num[((size_t)dst * H + h) * 32 + lane],
            w * v0[((size_t)src * H + h) * 32 + lane]);
  atomicAdd(&den[((size_t)dst * H + h) * 32 + lane], w);
}

// v = num/den (guarded); r = relu(v0+v); LN over (H,HD)=256 per node -> f16 (H,NN,32)
__global__ void k_vcomb(const float* __restrict__ num, const float* __restrict__ den,
                        const float* __restrict__ v0, const float* __restrict__ gam,
                        const float* __restrict__ bet, f16_t* __restrict__ vf) {
  __shared__ float s1[256], s2[256];
  int b = blockIdx.x, tid = threadIdx.x;
  int h = tid >> 5, d = tid & 31;
  size_t idx = ((size_t)b * H + h) * 32 + d;
  float dn = den[idx];
  float vv = dn > 0.f ? num[idx] / dn : 0.f;
  float r  = fmaxf(v0[idx] + vv, 0.f);
  s1[tid] = r; s2[tid] = r * r; __syncthreads();
  for (int st = 128; st; st >>= 1) {
    if (tid < st) { s1[tid] += s1[tid + st]; s2[tid] += s2[tid + st]; }
    __syncthreads();
  }
  float m   = s1[0] * (1.f / 256.f);
  float var = s2[0] * (1.f / 256.f) - m * m;
  float yn  = (r - m) * rsqrtf(var + 1e-5f) * gam[tid] + bet[tid];
  vf[((size_t)h * NN + b) * 32 + d] = (f16_t)yn;
}

// gates (H,NN,128): o,g,f,i -> update h,c; out = relu(o + xx)
__global__ void k_cell_gate(const float* __restrict__ gates, const float* __restrict__ xx,
                            int rs, int hs, float* __restrict__ hst, float* __restrict__ cst,
                            float* __restrict__ outb) {
  int i = blockIdx.x * blockDim.x + threadIdx.x;   // NN*H*32
  if (i >= NN * H * 32) return;
  int d = i & 31, h = (i >> 5) & 7, b = i >> 8;
  size_t g0 = ((size_t)h * NN + b) * 128;
  float o  = gates[g0 + d];
  float gg = gates[g0 + 32 + d];
  float f  = gates[g0 + 64 + d];
  float ii = gates[g0 + 96 + d];
  size_t si = ((size_t)b * H + h) * 32 + d;
  float cn = sigm(f) * cst[si] + sigm(ii) * tanhf(gg);
  float hn = sigm(o) * tanhf(cn);
  cst[si] = cn; hst[si] = hn;
  outb[si] = fmaxf(o + xx[(size_t)b * rs + h * hs + d], 0.f);
}

// --------------------------- head-mix / sampling ---------------------------
__global__ void k_h2pack(const float* __restrict__ hp, const float* __restrict__ hm,
                         f16_t* __restrict__ h2h) {
  int i = blockIdx.x * blockDim.x + threadIdx.x;   // NN*H*64
  if (i >= NN * H * 64) return;
  int j = i & 63, row = i >> 6;
  const float* s = (j & 1) ? hm : hp;
  h2h[i] = (f16_t)s[(size_t)row * 32 + (j >> 1)];
}

__global__ void k_ms(const float* __restrict__ h2o, const float* __restrict__ pe,
                     const int* __restrict__ batch, float* __restrict__ ms) {
  int i = blockIdx.x * blockDim.x + threadIdx.x;   // NN*512
  if (i >= NN * 512) return;
  int b = i >> 9, r = i & 511, s = r >> 8, j = r & 255;
  int h = j >> 5, d = j & 31;
  float v = h2o[((size_t)b * H + h) * 64 + s * 32 + d] * pe[b * 32 + d];
  atomicAdd(&ms[((size_t)(s * NB + batch[b])) * 256 + j], v);
}

__global__ void k_lnms(const float* __restrict__ ms, const float* __restrict__ mg,
                       const float* __restrict__ mb, const float* __restrict__ sg,
                       const float* __restrict__ sb, float* __restrict__ muln,
                       float* __restrict__ sigln) {
  int wid = threadIdx.x >> 5, lane = threadIdx.x & 31;
  int gw = blockIdx.x * 8 + wid;                  // 2*NB*H = 128 waves
  if (gw >= 2 * NB * H) return;
  int s = gw >> 6, nb = (gw >> 3) & 7, h = gw & 7;
  float v  = ms[((size_t)(s * NB + nb) * H + h) * 32 + lane];
  float m  = wsum(v) * (1.f / 32.f);
  float df = v - m;
  float var = wsum(df * df) * (1.f / 32.f);
  const float* g  = s ? sg : mg;
  const float* be = s ? sb : mb;
  float yn = df * rsqrtf(var + 1e-5f) * g[lane] + be[lane];
  (s ? sigln : muln)[((size_t)nb * H + h) * 32 + lane] = yn;
}

__global__ void k_seqin(const float* __restrict__ muln, const float* __restrict__ sigln,
                        const float* __restrict__ eps, const float* __restrict__ pe,
                        const int* __restrict__ batch, f16_t* __restrict__ seq) {
  long long i = (long long)blockIdx.x * blockDim.x + threadIdx.x;  // PP*NN*256
  if (i >= (long long)PP * NN * 256) return;
  int j = (int)(i & 255);
  int b = (int)((i >> 8) % NN);
  int p = (int)(i / ((long long)NN * 256));
  int h = j >> 5, d = j & 31;
  int nb = batch[b];
  size_t gi = ((size_t)nb * H + h) * 32 + d;
  float ov = muln[gi] + __expf(sigln[gi]) *
             eps[(((size_t)p * NB + nb) * H + h) * 32 + d];
  seq[i] = (f16_t)(ov * pe[b * 32 + d]);
}

// ----------------------------- particle LSTM -------------------------------
__global__ void k_big_gate(const float* __restrict__ gates, float* __restrict__ c,
                           f16_t* __restrict__ h16) {
  long long i = (long long)blockIdx.x * blockDim.x + threadIdx.x;  // PP*NN*256
  if (i >= (long long)PP * NN * 256) return;
  int d = (int)(i & 255);
  long long row = i >> 8;
  size_t g0 = (size_t)row * G4;
  float ii = gates[g0 + d];
  float ff = gates[g0 + 256 + d];
  float gg = gates[g0 + 512 + d];
  float oo = gates[g0 + 768 + d];
  float cn = sigm(ff) * c[i] + sigm(ii) * tanhf(gg);
  c[i] = cn;
  h16[i] = (f16_t)(sigm(oo) * tanhf(cn));
}

// leaky -> LN(32) -> fc2 (32x31), wave per row, write transposed output slot
__global__ void k_fc2(const float* __restrict__ y, const float* __restrict__ g,
                      const float* __restrict__ be, const float* __restrict__ W2,
                      const float* __restrict__ b2, float* __restrict__ out, int step) {
  int wid = threadIdx.x >> 5, lane = threadIdx.x & 31;
  int row = blockIdx.x * 8 + wid;
  if (row >= PP * NN) return;
  float v  = lrelu(y[(size_t)row * 32 + lane]);
  float m  = wsum(v) * (1.f / 32.f);
  float df = v - m;
  float var = wsum(df * df) * (1.f / 32.f);
  float yn = df * rsqrtf(var + 1e-5f) * g[lane] + be[lane];
  float acc = (lane < 31) ? b2[lane] : 0.f;
  for (int dd = 0; dd < 32; ++dd) {
    float yv = __shfl(yn, dd, 32);
    if (lane < 31) acc += yv * W2[dd * 31 + lane];
  }
  if (lane < 31) out[((size_t)row * W + step) * LAM + lane] = acc;
}

// ---------------------------------------------------------------------------
static inline void gemm(hipStream_t st, const f16_t* A, const f16_t* BT,
                        const float* bias, const float* add, float* C,
                        int M, int N, int K, long long sA, long long sB,
                        long long sC, int Z, int flags) {
  dim3 grid((N + BN - 1) / BN, (M + BM - 1) / BM, Z);
  k_wmma_gemm<<<grid, 256, 0, st>>>(A, BT, bias, add, C, M, N, K, sA, sB, sC, flags);
}
static inline int blks(long long n) { return (int)((n + 255) / 256); }

extern "C" void kernel_launch(void* const* d_in, const int* in_sizes, int n_in,
                              void* d_out, int out_size, void* d_ws, size_t ws_size,
                              hipStream_t stream) {
  (void)in_sizes; (void)n_in; (void)out_size; (void)ws_size;
  const float* x      = (const float*)d_in[0];
  const float* time_t = (const float*)d_in[1];
  const float* pos    = (const float*)d_in[2];
  const float* eps    = (const float*)d_in[3];
  const float* emb_W  = (const float*)d_in[4];
  const float* emb_b  = (const float*)d_in[5];
  const float* pe_W1  = (const float*)d_in[6];
  const float* pe_b1  = (const float*)d_in[7];
  const float* pe_g1  = (const float*)d_in[8];
  const float* pe_be1 = (const float*)d_in[9];
  const float* pe_W2  = (const float*)d_in[10];
  const float* pe_b2  = (const float*)d_in[11];
  const float* Wqkv   = (const float*)d_in[12];
  const float* qkv_g  = (const float*)d_in[13];
  const float* qkv_b  = (const float*)d_in[14];
  const float* vn_g   = (const float*)d_in[15];
  const float* vn_b   = (const float*)d_in[16];
  const float* lin_W  = (const float*)d_in[17];
  const float* lin_b  = (const float*)d_in[18];
  const float* hlin_W = (const float*)d_in[19];
  const float* hlin_b = (const float*)d_in[20];
  const float* mu_g   = (const float*)d_in[21];
  const float* mu_b   = (const float*)d_in[22];
  const float* sig_g  = (const float*)d_in[23];
  const float* sig_b  = (const float*)d_in[24];
  const float* Wih    = (const float*)d_in[25];
  const float* Whh    = (const float*)d_in[26];
  const float* bih    = (const float*)d_in[27];
  const float* bhh    = (const float*)d_in[28];
  const float* fc_W1  = (const float*)d_in[29];
  const float* fc_b1  = (const float*)d_in[30];
  const float* fc_g   = (const float*)d_in[31];
  const float* fc_be  = (const float*)d_in[32];
  const float* fc_W2  = (const float*)d_in[33];
  const float* fc_b2  = (const float*)d_in[34];
  const int*   ei     = (const int*)d_in[35];
  const int*   batch  = (const int*)d_in[36];
  float* out = (float*)d_out;

  // ---- workspace bump allocator (256B aligned) ----
  uint8_t* base = (uint8_t*)d_ws;
  size_t off = 0;
  auto alloc = [&](size_t bytes) -> void* {
    void* p = base + off;
    off = (off + bytes + 255) & ~(size_t)255;
    return p;
  };
  float* peW     = (float*)alloc((size_t)NN * 32 * 4);
  float* xe      = (float*)alloc((size_t)W * NN * 32 * 4);
  float* biasC   = (float*)alloc((size_t)G4 * 4);
  float* traw    = (float*)alloc((size_t)H * NN * 96 * 4);
  float* qb      = (float*)alloc((size_t)NN * HH * 4);
  float* kb      = (float*)alloc((size_t)NN * HH * 4);
  float* v0b     = (float*)alloc((size_t)NN * HH * 4);
  float* numb    = (float*)alloc((size_t)NN * HH * 4);
  float* denb    = (float*)alloc((size_t)NN * HH * 4);
  float* gatesc  = (float*)alloc((size_t)H * NN * 128 * 4);
  float* hp      = (float*)alloc((size_t)NN * HH * 4);
  float* cp      = (float*)alloc((size_t)NN * HH * 4);
  float* hm      = (float*)alloc((size_t)NN * HH * 4);
  float* cm      = (float*)alloc((size_t)NN * HH * 4);
  float* opb     = (float*)alloc((size_t)NN * HH * 4);
  float* omb     = (float*)alloc((size_t)NN * HH * 4);
  float* h2o     = (float*)alloc((size_t)NN * H * 64 * 4);
  float* msb     = (float*)alloc((size_t)2 * NB * 256 * 4);
  float* muln    = (float*)alloc((size_t)NB * 256 * 4);
  float* sigln   = (float*)alloc((size_t)NB * 256 * 4);
  float* xproj   = (float*)alloc((size_t)PP * NN * G4 * 4);
  float* gbig    = (float*)alloc((size_t)PP * NN * G4 * 4);
  float* cbig    = (float*)alloc((size_t)PP * NN * HH * 4);
  float* yraw    = (float*)alloc((size_t)PP * NN * 32 * 4);
  f16_t* qkvWh   = (f16_t*)alloc((size_t)4 * H * 96 * 96 * 2);
  f16_t* linWh   = (f16_t*)alloc((size_t)4 * 128 * 32 * 2);
  f16_t* hlinWh  = (f16_t*)alloc((size_t)64 * 64 * 2);
  f16_t* wihH    = (f16_t*)alloc((size_t)G4 * HH * 2);
  f16_t* whhH    = (f16_t*)alloc((size_t)G4 * HH * 2);
  f16_t* fcw1H   = (f16_t*)alloc((size_t)32 * HH * 2);
  f16_t* xhcH    = (f16_t*)alloc((size_t)H * NN * 96 * 2);
  f16_t* vfinH   = (f16_t*)alloc((size_t)NN * HH * 2);
  f16_t* h2H     = (f16_t*)alloc((size_t)NN * H * 64 * 2);
  f16_t* seqH    = (f16_t*)alloc((size_t)PP * NN * HH * 2);
  f16_t* hbigH   = (f16_t*)alloc((size_t)PP * NN * HH * 2);

  // ---- weight conversion (BT layout = [N][K] f16) ----
  k_f32h <<<blks(4LL*H*96*96), 256, 0, stream>>>(Wqkv,   qkvWh, 4LL*H*96*96);
  k_f32hT<<<blks(4LL*32*128),  256, 0, stream>>>(lin_W,  linWh, 32, 128, 4LL*32*128);
  k_f32hT<<<blks(64LL*64),     256, 0, stream>>>(hlin_W, hlinWh, 64, 64, 64LL*64);
  k_f32h <<<blks((long long)G4*HH), 256, 0, stream>>>(Wih, wihH, (long long)G4*HH);
  k_f32h <<<blks((long long)G4*HH), 256, 0, stream>>>(Whh, whhH, (long long)G4*HH);
  k_f32hT<<<blks((long long)HH*32), 256, 0, stream>>>(fc_W1, fcw1H, HH, 32, (long long)HH*32);
  k_add2 <<<blks(G4), 256, 0, stream>>>(bih, bhh, biasC, G4);

  // ---- positional encoding + embedding ----
  k_pe   <<<(NN + 7) / 8, 256, 0, stream>>>(pos, pe_W1, pe_b1, pe_g1, pe_be1, pe_W2, pe_b2, peW);
  k_embed<<<blks((long long)W*NN*32), 256, 0, stream>>>(x, time_t, emb_W, emb_b, xe);

  // ---- init graph-cell states ----
  k_zero<<<blks((long long)NN*HH), 256, 0, stream>>>(hp, (long long)NN*HH);
  k_zero<<<blks((long long)NN*HH), 256, 0, stream>>>(cp, (long long)NN*HH);
  k_zero<<<blks((long long)NN*HH), 256, 0, stream>>>(hm, (long long)NN*HH);
  k_zero<<<blks((long long)NN*HH), 256, 0, stream>>>(cm, (long long)NN*HH);

  auto run_cell = [&](int idx, const float* xx, int rs, int hs,
                      float* hst, float* cst, float* outb) {
    k_pack_xhc<<<blks((long long)H*NN*96), 256, 0, stream>>>(xhcH, xx, rs, hs, hst, cst);
    gemm(stream, xhcH, qkvWh + (size_t)idx * H * 96 * 96, nullptr, nullptr, traw,
         NN, 96, 96, (long long)NN * 96, 96LL * 96, (long long)NN * 96, H, 0);
    k_qkv_norm<<<NN, 256, 0, stream>>>(traw, qkv_g + idx * 768, qkv_b + idx * 768, qb, kb, v0b);
    k_zero<<<blks((long long)NN*HH), 256, 0, stream>>>(numb, (long long)NN*HH);
    k_zero<<<blks((long long)NN*HH), 256, 0, stream>>>(denb, (long long)NN*HH);
    k_attn<<<blks((long long)EE*H*32), 256, 0, stream>>>(qb, kb, v0b, peW, ei, numb, denb);
    k_vcomb<<<NN, 256, 0, stream>>>(numb, denb, v0b, vn_g + idx * 256, vn_b + idx * 256, vfinH);
    gemm(stream, vfinH, linWh + (size_t)idx * 128 * 32, lin_b + idx * 128, nullptr, gatesc,
         NN, 128, 32, (long long)NN * 32, 0LL, (long long)NN * 128, H, 1);
    k_cell_gate<<<blks((long long)NN*HH), 256, 0, stream>>>(gatesc, xx, rs, hs, hst, cst, outb);
  };

  // ---- bidirectional graph BiLSTM scan: 6 steps x 2 layers x 2 dirs ----
  for (int t = 0; t < W; ++t) {
    run_cell(0, xe + (size_t)t * NN * 32,           32, 0,  hp, cp, opb);
    run_cell(1, xe + (size_t)(W - 1 - t) * NN * 32, 32, 0,  hm, cm, omb);
    run_cell(2, opb, 256, 32, hp, cp, opb);
    run_cell(3, omb, 256, 32, hm, cm, omb);
  }

  // ---- head mix, segment sums, latent sampling ----
  k_h2pack<<<blks((long long)NN*H*64), 256, 0, stream>>>(hp, hm, h2H);
  gemm(stream, h2H, hlinWh, hlin_b, nullptr, h2o, NN * H, 64, 64, 0, 0, 0, 1, 1);
  k_zero<<<blks(2LL*NB*256), 256, 0, stream>>>(msb, 2LL * NB * 256);
  k_ms  <<<blks((long long)NN*512), 256, 0, stream>>>(h2o, peW, batch, msb);
  k_lnms<<<16, 256, 0, stream>>>(msb, mu_g, mu_b, sig_g, sig_b, muln, sigln);
  k_seqin<<<blks((long long)PP*NN*HH), 256, 0, stream>>>(muln, sigln, eps, peW, batch, seqH);

  // ---- particle LSTM: xproj once, 6 recurrent WMMA steps, fused fc tail ----
  gemm(stream, seqH, wihH, biasC, nullptr, xproj, PP * NN, G4, HH, 0, 0, 0, 1, 1);
  k_zero<<<blks((long long)PP*NN*HH/2), 256, 0, stream>>>((float*)hbigH, (long long)PP*NN*HH/2);
  k_zero<<<blks((long long)PP*NN*HH),   256, 0, stream>>>(cbig, (long long)PP*NN*HH);
  for (int s = 0; s < W; ++s) {
    gemm(stream, hbigH, whhH, nullptr, xproj, gbig, PP * NN, G4, HH, 0, 0, 0, 1, 2);
    k_big_gate<<<blks((long long)PP*NN*HH), 256, 0, stream>>>(gbig, cbig, hbigH);
    gemm(stream, hbigH, fcw1H, fc_b1, nullptr, yraw, PP * NN, 32, HH, 0, 0, 0, 1, 1);
    k_fc2<<<(PP * NN + 7) / 8, 256, 0, stream>>>(yraw, fc_g, fc_be, fc_W2, fc_b2, out, s);
  }

  // ---- tuple tail: x passthrough + musig ----
  k_copy<<<blks((long long)NN*W*LAM), 256, 0, stream>>>(x, out + (size_t)PP * NN * W * LAM,
                                                        (long long)NN * W * LAM);
  k_copy<<<blks(2LL*NB*256), 256, 0, stream>>>(msb,
      out + (size_t)PP * NN * W * LAM + (size_t)NN * W * LAM, 2LL * NB * 256);
}
